// KernelAttention_12532714569864
// MI455X (gfx1250) — matrix operand order repaired and checked by reading, not compiled, not run on hardware
//
#include <hip/hip_runtime.h>
#include <hip/hip_bf16.h>
#include <math.h>

typedef __attribute__((ext_vector_type(16))) _Float16 v16h;
typedef __attribute__((ext_vector_type(8)))  float    v8f;

union U16H { uint4 u[2]; v16h h; };

#define WMMA(A,B,C) __builtin_amdgcn_wmma_f32_16x16x32_f16(false,(A),false,(B),(short)0,(C),false,false)

// ---- constants ----
#define BQ   2048
#define BK   2048
#define DIN  128
#define HD   64
#define NB   4
#define KCH  4                      // K-split chunks for pass 1
// exp(-d2/8) = exp2(d2 * SC)
#define SCL  (-0.18033688011112042f)

// A fragment (16x32 f16) from f16 row-major matrix; row stride ld (elements).
__device__ __forceinline__ v16h load_a16(const _Float16* base, int ld, int row0, int c0,
                                         int m15, int hf) {
    const char* p = (const char*)(base + (size_t)(row0 + m15) * ld) + (size_t)(c0 + 16 * hf) * 2;
    U16H t;
    t.u[0] = *(const uint4*)p;          // K = c0+16h .. +7
    t.u[1] = *(const uint4*)(p + 16);   // K = c0+16h+8 .. +7
    return t.h;
}

// A fragment from f32 row-major source, converting to f16.
__device__ __forceinline__ v16h load_a32(const float* base, int ld, int row0, int c0,
                                         int m15, int hf) {
    const float* p = base + (size_t)(row0 + m15) * ld + c0 + 16 * hf;
    v16h a;
#pragma unroll
    for (int i = 0; i < 8; ++i) a[i] = (_Float16)p[i];
#pragma unroll
    for (int i = 0; i < 8; ++i) a[8 + i] = (_Float16)p[8 + i];
    return a;
}

// ---------------- weight transpose/convert ----------------
__global__ __launch_bounds__(256) void setup_weights(
    const float* __restrict__ Wq, const float* __restrict__ Wk,
    const float* __restrict__ Wv, const float* __restrict__ Wo,
    _Float16* __restrict__ WqT, _Float16* __restrict__ WkT,
    _Float16* __restrict__ WvT, _Float16* __restrict__ WoT) {
    int t = blockIdx.x * blockDim.x + threadIdx.x;
    int stride = gridDim.x * blockDim.x;
    for (int i = t; i < DIN * HD; i += stride) {   // [128][64] -> [64][128]
        int d = i >> 6, h = i & 63;
        WqT[h * DIN + d] = (_Float16)Wq[i];
        WkT[h * DIN + d] = (_Float16)Wk[i];
        WvT[h * DIN + d] = (_Float16)Wv[i];
    }
    for (int i = t; i < HD * HD; i += stride) {    // [64][64] -> [64][64]^T
        int hi = i >> 6, ho = i & 63;
        WoT[ho * HD + hi] = (_Float16)Wo[i];
    }
}

// ---------------- projections: q/k/v = X*W + b ----------------
// grid = 3 * NB * (2048/16), block = 32 (one wave per 16-row tile)
__global__ __launch_bounds__(32) void proj_kernel(
    const float* __restrict__ qs, const float* __restrict__ ks, const float* __restrict__ vs,
    const float* __restrict__ bq, const float* __restrict__ bk, const float* __restrict__ bv,
    const _Float16* __restrict__ WqT, const _Float16* __restrict__ WkT,
    const _Float16* __restrict__ WvT,
    _Float16* __restrict__ q16, _Float16* __restrict__ k16, _Float16* __restrict__ vT,
    float* __restrict__ q2, float* __restrict__ k2) {
    const int lane = threadIdx.x;
    const int m15 = lane & 15, hf = lane >> 4;
    const int id = blockIdx.x;
    const int type = id >> 9;          // 512 tiles per matrix
    const int rem = id & 511;
    const int b = rem >> 7;
    const int row0 = (rem & 127) * 16;

    const float* src; const float* bias; const _Float16* WT;
    if (type == 0)      { src = qs; bias = bq; WT = WqT; }
    else if (type == 1) { src = ks; bias = bk; WT = WkT; }
    else                { src = vs; bias = bv; WT = WvT; }
    const float* srcb = src + (size_t)b * BQ * DIN;

    v8f acc[4] = {};
#pragma unroll
    for (int cc = 0; cc < 4; ++cc) {
        v16h a = load_a32(srcb, DIN, row0, cc * 32, m15, hf);
#pragma unroll
        for (int nt = 0; nt < 4; ++nt) {
            const _Float16* pb = WT + (size_t)(nt * 16 + m15) * DIN + cc * 32 + 16 * hf;
            v16h bf = *(const v16h*)pb;
            acc[nt] = WMMA(a, bf, acc[nt]);
        }
    }
#pragma unroll
    for (int nt = 0; nt < 4; ++nt) {
        float bb = bias[nt * 16 + m15];
#pragma unroll
        for (int r = 0; r < 8; ++r) acc[nt][r] += bb;
    }

    if (type < 2) {
        // row norms (reduce across the 16 lanes of each half-wave)
        float part[8];
#pragma unroll
        for (int r = 0; r < 8; ++r) {
            part[r] = 0.f;
#pragma unroll
            for (int nt = 0; nt < 4; ++nt) part[r] += acc[nt][r] * acc[nt][r];
        }
#pragma unroll
        for (int r = 0; r < 8; ++r) {
            part[r] += __shfl_xor(part[r], 1, 32);
            part[r] += __shfl_xor(part[r], 2, 32);
            part[r] += __shfl_xor(part[r], 4, 32);
            part[r] += __shfl_xor(part[r], 8, 32);
        }
        float* dn = (type == 0 ? q2 : k2) + b * BQ + row0;
        if (m15 == 0) {
#pragma unroll
            for (int r = 0; r < 8; ++r) dn[r + 8 * hf] = part[r];
        }
        _Float16* d16 = (type == 0 ? q16 : k16) + ((size_t)b * BQ + row0) * HD;
#pragma unroll
        for (int nt = 0; nt < 4; ++nt)
#pragma unroll
            for (int r = 0; r < 8; ++r)
                d16[(size_t)(r + 8 * hf) * HD + nt * 16 + m15] = (_Float16)acc[nt][r];
    } else {
        // transpose v -> vT[b][h][n] through LDS
        __shared__ alignas(32) _Float16 vt[64][16];
#pragma unroll
        for (int nt = 0; nt < 4; ++nt)
#pragma unroll
            for (int r = 0; r < 8; ++r)
                vt[nt * 16 + m15][r + 8 * hf] = (_Float16)acc[nt][r];
        __syncthreads();
        _Float16* dv = vT + (size_t)b * HD * BK + row0;
#pragma unroll
        for (int w = 0; w < 2; ++w) {
            int hh = lane + 32 * w;
            const uint4* s = (const uint4*)&vt[hh][0];
            uint4* d = (uint4*)(dv + (size_t)hh * BK);
            d[0] = s[0];
            d[1] = s[1];
        }
    }
}

// ---------------- pass 1a: K-split partial rowsums + partial ctx = P*V ----------------
// grid = NB * 128 * KCH, block = 32 (one wave: 16 q rows x 512 keys)
__global__ __launch_bounds__(32) void attn_part_kernel(
    const _Float16* __restrict__ q16, const _Float16* __restrict__ k16,
    const _Float16* __restrict__ vT,
    const float* __restrict__ q2, const float* __restrict__ k2,
    float* __restrict__ rsP, float* __restrict__ ctxP) {
    const int lane = threadIdx.x;
    const int m15 = lane & 15, hf = lane >> 4;
    const int id = blockIdx.x;
    const int b = id >> 9;                 // 128*KCH blocks per batch
    const int rem = id & 511;
    const int row0 = (rem >> 2) * 16;
    const int ch = rem & 3;
    const int jb = ch * (BK / KCH);

    const _Float16* qb = q16 + (size_t)b * BQ * HD;
    const _Float16* kb = k16 + (size_t)b * BK * HD;
    const _Float16* vb = vT + (size_t)b * HD * BK;

    const v16h aq0 = load_a16(qb, HD, row0, 0, m15, hf);
    const v16h aq1 = load_a16(qb, HD, row0, 32, m15, hf);

    float q2r[8];
#pragma unroll
    for (int r = 0; r < 8; ++r) q2r[r] = q2[b * BQ + row0 + r + 8 * hf];

    float rs[8] = {};
    v8f cacc[4] = {};
    __shared__ alignas(16) _Float16 Pt[2][16][40];   // double-buffered P staging

    for (int j0 = jb; j0 < jb + BK / KCH; j0 += 32) {
        const int pf = (j0 >> 5) & 1;
        __builtin_prefetch(kb + (size_t)(j0 + 32 + m15) * HD, 0, 3);
        __builtin_prefetch(vb + (size_t)m15 * BK + j0 + 32, 0, 3);
#pragma unroll
        for (int st = 0; st < 2; ++st) {
            const int jj = j0 + st * 16;
            const _Float16* kr = kb + (size_t)(jj + m15) * HD + 16 * hf;
            v16h bk0 = *(const v16h*)kr;
            v16h bk1 = *(const v16h*)(kr + 32);
            v8f s = {};
            s = WMMA(aq0, bk0, s);
            s = WMMA(aq1, bk1, s);
            const float k2v = k2[b * BK + jj + m15];
#pragma unroll
            for (int r = 0; r < 8; ++r) {
                float d2 = q2r[r] + k2v - 2.0f * s[r];
                float pv = exp2f(d2 * SCL);
                rs[r] += pv;
                Pt[pf][r + 8 * hf][st * 16 + m15] = (_Float16)pv;
            }
        }
        __syncthreads();
        // P tile as A fragment (C-layout -> A-layout pivot through LDS)
        U16H t;
        {
            const char* pr = (const char*)&Pt[pf][m15][0];
            t.u[0] = *(const uint4*)(pr + 32 * hf);
            t.u[1] = *(const uint4*)(pr + 16 + 32 * hf);
        }
        const v16h ap = t.h;
#pragma unroll
        for (int nt = 0; nt < 4; ++nt) {
            const _Float16* pv = vb + (size_t)(nt * 16 + m15) * BK + j0 + 16 * hf;
            v16h bv = *(const v16h*)pv;
            cacc[nt] = WMMA(ap, bv, cacc[nt]);
        }
        // no trailing barrier: next iteration uses the other buffer
    }

    // partial row sums within each half-wave
#pragma unroll
    for (int r = 0; r < 8; ++r) {
        rs[r] += __shfl_xor(rs[r], 1, 32);
        rs[r] += __shfl_xor(rs[r], 2, 32);
        rs[r] += __shfl_xor(rs[r], 4, 32);
        rs[r] += __shfl_xor(rs[r], 8, 32);
    }
    if (m15 == 0) {
#pragma unroll
        for (int r = 0; r < 8; ++r)
            rsP[((size_t)(ch * NB + b)) * BQ + row0 + r + 8 * hf] = rs[r];
    }
    float* cp = ctxP + (((size_t)(ch * NB + b)) * BQ + row0) * HD;
#pragma unroll
    for (int nt = 0; nt < 4; ++nt)
#pragma unroll
        for (int r = 0; r < 8; ++r)
            cp[(size_t)(r + 8 * hf) * HD + nt * 16 + m15] = cacc[nt][r];
}

// ---------------- pass 1b: combine partials, out = (ctx/rowsum)*Wo + bo ----------------
// grid = NB * 128, block = 32
__global__ __launch_bounds__(32) void ctx_out_kernel(
    const float* __restrict__ rsP, const float* __restrict__ ctxP,
    const _Float16* __restrict__ WoT, const float* __restrict__ bo,
    float* __restrict__ out, float* __restrict__ invR) {
    const int lane = threadIdx.x;
    const int m15 = lane & 15, hf = lane >> 4;
    const int b = blockIdx.x >> 7;
    const int row0 = (blockIdx.x & 127) * 16;

    float rs[8] = {};
#pragma unroll
    for (int ch = 0; ch < KCH; ++ch)
#pragma unroll
        for (int r = 0; r < 8; ++r)
            rs[r] += rsP[((size_t)(ch * NB + b)) * BQ + row0 + r + 8 * hf];
    float inv[8];
#pragma unroll
    for (int r = 0; r < 8; ++r) inv[r] = 1.0f / rs[r];
    if (m15 == 0) {
#pragma unroll
        for (int r = 0; r < 8; ++r) invR[b * BQ + row0 + r + 8 * hf] = inv[r];
    }

    float c[4][8] = {};
#pragma unroll
    for (int ch = 0; ch < KCH; ++ch) {
        const float* cp = ctxP + (((size_t)(ch * NB + b)) * BQ + row0) * HD;
#pragma unroll
        for (int nt = 0; nt < 4; ++nt)
#pragma unroll
            for (int r = 0; r < 8; ++r)
                c[nt][r] += cp[(size_t)(r + 8 * hf) * HD + nt * 16 + m15];
    }

    __shared__ alignas(16) _Float16 Ct[16][72];
#pragma unroll
    for (int nt = 0; nt < 4; ++nt)
#pragma unroll
        for (int r = 0; r < 8; ++r)
            Ct[r + 8 * hf][nt * 16 + m15] = (_Float16)(c[nt][r] * inv[r]);
    __syncthreads();

    v16h ac[2];
#pragma unroll
    for (int cc = 0; cc < 2; ++cc) {
        const char* pr = (const char*)&Ct[m15][0];
        U16H t2;
        t2.u[0] = *(const uint4*)(pr + cc * 64 + 32 * hf);
        t2.u[1] = *(const uint4*)(pr + cc * 64 + 16 + 32 * hf);
        ac[cc] = t2.h;
    }
    float* ob = out + ((size_t)b * BQ + row0) * HD;
#pragma unroll
    for (int nt = 0; nt < 4; ++nt) {
        v8f o = {};
#pragma unroll
        for (int cc = 0; cc < 2; ++cc) {
            const _Float16* pw = WoT + (size_t)(nt * 16 + m15) * HD + cc * 32 + 16 * hf;
            v16h bw = *(const v16h*)pw;
            o = WMMA(ac[cc], bw, o);
        }
        float bb = bo[nt * 16 + m15];
#pragma unroll
        for (int r = 0; r < 8; ++r)
            ob[(size_t)(r + 8 * hf) * HD + nt * 16 + m15] = o[r] + bb;
    }
}

// ---------------- pass 2: recompute S, write normalized attn (bandwidth phase) ----------------
// grid = NB * 128 * 8 (each wave: 16 q rows x 256 keys), block = 32
__global__ __launch_bounds__(32) void attn_write_kernel(
    const _Float16* __restrict__ q16, const _Float16* __restrict__ k16,
    const float* __restrict__ q2, const float* __restrict__ k2,
    const float* __restrict__ invR, float* __restrict__ attn) {
    const int lane = threadIdx.x;
    const int m15 = lane & 15, hf = lane >> 4;
    const int id = blockIdx.x;
    const int b = id >> 10;                 // 128*8 blocks per batch
    const int rem = id & 1023;
    const int row0 = (rem >> 3) * 16;
    const int jb = (rem & 7) * 256;

    const _Float16* qb = q16 + (size_t)b * BQ * HD;
    const _Float16* kb = k16 + (size_t)b * BK * HD;

    const v16h aq0 = load_a16(qb, HD, row0, 0, m15, hf);
    const v16h aq1 = load_a16(qb, HD, row0, 32, m15, hf);

    float q2r[8], inv[8];
#pragma unroll
    for (int r = 0; r < 8; ++r) {
        q2r[r] = q2[b * BQ + row0 + r + 8 * hf];
        inv[r] = invR[b * BQ + row0 + r + 8 * hf];
    }

    __shared__ alignas(16) float St[2][16][32];    // double-buffered staging

    for (int j0 = jb; j0 < jb + 256; j0 += 32) {
        const int pf = (j0 >> 5) & 1;
        __builtin_prefetch(kb + (size_t)(j0 + 32 + m15) * HD, 0, 3);
#pragma unroll
        for (int st = 0; st < 2; ++st) {
            const int jj = j0 + st * 16;
            const _Float16* kr = kb + (size_t)(jj + m15) * HD + 16 * hf;
            v16h bk0 = *(const v16h*)kr;
            v16h bk1 = *(const v16h*)(kr + 32);
            v8f s = {};
            s = WMMA(aq0, bk0, s);
            s = WMMA(aq1, bk1, s);
            const float k2v = k2[b * BK + jj + m15];
#pragma unroll
            for (int r = 0; r < 8; ++r) {
                float d2 = q2r[r] + k2v - 2.0f * s[r];
                St[pf][r + 8 * hf][st * 16 + m15] = exp2f(d2 * SCL) * inv[r];
            }
        }
        __syncthreads();
        // each lane writes a contiguous 64B span; lane pairs form 128B per attn row
        const int rr = lane >> 1;
        const int c0 = (lane & 1) * 16;
        const float4* s4 = (const float4*)&St[pf][rr][c0];
        float4 x0 = s4[0], x1 = s4[1], x2 = s4[2], x3 = s4[3];
        float4* d4 = (float4*)(attn + ((size_t)b * BQ + row0 + rr) * BK + j0 + c0);
        d4[0] = x0; d4[1] = x1; d4[2] = x2; d4[3] = x3;
        // no trailing barrier: next iteration uses the other buffer
    }
}

extern "C" void kernel_launch(void* const* d_in, const int* in_sizes, int n_in,
                              void* d_out, int out_size, void* d_ws, size_t ws_size,
                              hipStream_t stream) {
    const float* qs = (const float*)d_in[0];
    const float* ks = (const float*)d_in[1];
    const float* vs = (const float*)d_in[2];
    const float* Wq = (const float*)d_in[3];
    const float* bq = (const float*)d_in[4];
    const float* Wk = (const float*)d_in[5];
    const float* bk = (const float*)d_in[6];
    const float* Wv = (const float*)d_in[7];
    const float* bv = (const float*)d_in[8];
    const float* Wo = (const float*)d_in[9];
    const float* bo = (const float*)d_in[10];

    char* ws = (char*)d_ws;
    _Float16* q16 = (_Float16*)(ws + 0);
    _Float16* k16 = (_Float16*)(ws + 1048576);
    _Float16* vT  = (_Float16*)(ws + 2097152);
    float*    q2  = (float*)(ws + 3145728);
    float*    k2  = (float*)(ws + 3178496);
    float*    invR= (float*)(ws + 3211264);
    _Float16* WqT = (_Float16*)(ws + 3244032);
    _Float16* WkT = (_Float16*)(ws + 3260416);
    _Float16* WvT = (_Float16*)(ws + 3276800);
    _Float16* WoT = (_Float16*)(ws + 3293184);
    float*    rsP = (float*)(ws + 3301376);               // KCH*NB*BQ f32 = 128KB
    float*    ctxP= (float*)(ws + 3432448);               // KCH*NB*BQ*HD f32 = 8MB

    float* out = (float*)d_out;
    float* attn = out + (size_t)NB * BQ * HD;

    setup_weights<<<32, 256, 0, stream>>>(Wq, Wk, Wv, Wo, WqT, WkT, WvT, WoT);
    proj_kernel<<<3 * NB * (BQ / 16), 32, 0, stream>>>(qs, ks, vs, bq, bk, bv,
                                                       WqT, WkT, WvT,
                                                       q16, k16, vT, q2, k2);
    attn_part_kernel<<<NB * (BQ / 16) * KCH, 32, 0, stream>>>(q16, k16, vT, q2, k2,
                                                              rsP, ctxP);
    ctx_out_kernel<<<NB * (BQ / 16), 32, 0, stream>>>(rsP, ctxP, WoT, bo, out, invR);
    attn_write_kernel<<<NB * (BQ / 16) * 8, 32, 0, stream>>>(q16, k16, q2, k2, invR, attn);
}
